// RnCLoss_75557064671692
// MI455X (gfx1250) — compile-verified, matrix operand order checked by reading, not run on hardware
//
#include <hip/hip_runtime.h>
#include <hip/hip_bf16.h>
#include <math.h>

typedef __attribute__((ext_vector_type(16))) _Float16 v16h;
typedef __attribute__((ext_vector_type(8)))  float    v8f;

#define NROWS 4096        // n = 2*B
#define DDIM  256
#define NT    256
#define CHUNK (NROWS / NT)        // 16 elements per thread in row kernel
#define INV_SQRT_T 0.70710678118654752f   // 1/sqrt(TEMPERATURE=2)
#define LOSS_EPS 1e-8f
#define BIGF 3.0e38f

// ---------------------------------------------------------------------------
// Pass 1: normalize features (fold 1/(norm*sqrt(T)) in), pack labels.
// One block per row, 256 threads (= D).
// ---------------------------------------------------------------------------
__global__ void rnc_prep_kernel(const float* __restrict__ zi,
                                const float* __restrict__ zj,
                                const float* __restrict__ pi,
                                const float* __restrict__ pj,
                                _Float16* __restrict__ f16,
                                float* __restrict__ lab) {
    const int row = blockIdx.x;
    const int t   = threadIdx.x;
    const float* src = (row < NROWS / 2) ? (zi + (size_t)row * DDIM)
                                         : (zj + (size_t)(row - NROWS / 2) * DDIM);
    float v = src[t];

    __shared__ float red[NT];
    red[t] = v * v;
    __syncthreads();
    for (int s = NT / 2; s > 0; s >>= 1) {
        if (t < s) red[t] += red[t + s];
        __syncthreads();
    }
    const float scale = INV_SQRT_T / sqrtf(red[0]);
    f16[(size_t)row * DDIM + t] = (_Float16)(v * scale);

    if (t < 4) {
        const float* ls = (row < NROWS / 2) ? (pi + (size_t)row * 4)
                                            : (pj + (size_t)(row - NROWS / 2) * 4);
        lab[(size_t)row * 4 + t] = ls[t];
    }
}

// ---------------------------------------------------------------------------
// Pass 2: logits = F @ F^T (temperature already folded into F).
// 8 waves per block; each wave32 computes one 16x16 f32 tile with
// v_wmma_f32_16x16x32_f16, K-loop of 8 steps.
// ---------------------------------------------------------------------------
__global__ void rnc_gemm_kernel(const _Float16* __restrict__ F,
                                float* __restrict__ logits) {
    const int wave = threadIdx.x >> 5;
    const int lane = threadIdx.x & 31;
    const int tile = blockIdx.x * 8 + wave;
    const int mt   = tile >> 8;     // 256 column tiles (4096/16)
    const int nt   = tile & 255;

    const int sub  = lane & 15;
    const int koff = (lane >> 4) * 16;   // lanes 0-15: K 0..15, lanes 16-31: K 16..31

    const _Float16* pa = F + (size_t)(mt * 16 + sub) * DDIM + koff;
    const _Float16* pb = F + (size_t)(nt * 16 + sub) * DDIM + koff;

    v8f c = {};
#pragma unroll
    for (int k0 = 0; k0 < DDIM; k0 += 32) {
        __builtin_prefetch(pa + k0 + 32, 0, 0);   // global_prefetch_b8 next K slab
        __builtin_prefetch(pb + k0 + 32, 0, 0);
        v16h a = *(const v16h*)(pa + k0);
        v16h b = *(const v16h*)(pb + k0);
        // (neg_a, A, neg_b, B, c_mod, C, reuse_a, reuse_b)
        c = __builtin_amdgcn_wmma_f32_16x16x32_f16(false, a, false, b,
                                                   (short)0, c, false, false);
    }

    // C layout: VGPR r, lanes 0-15 -> M=r,   N=lane
    //                   lanes 16-31 -> M=r+8, N=lane-16
    const int colbase = nt * 16 + sub;
    const int rowbase = mt * 16 + ((lane >> 4) << 3);
#pragma unroll
    for (int r = 0; r < 8; ++r) {
        logits[(size_t)(rowbase + r) * NROWS + colbase] = c[r];
    }
}

// ---------------------------------------------------------------------------
// Pass 3: per-row ranked-softmax denominator.
// One block (256 threads) per row. LDS: ld keys + exp values (32KB) + scan.
// denom[j] = total - csum(exp sorted by ld)[rank(ld_j)-1], rank = lower_bound.
// ---------------------------------------------------------------------------
__global__ void rnc_row_kernel(const float* __restrict__ logits,
                               const float* __restrict__ lab,
                               float* __restrict__ rowloss) {
    const int row = blockIdx.x;
    const int t   = threadIdx.x;

    __shared__ float s_ld[NROWS];
    __shared__ float s_ex[NROWS];
    __shared__ float s_red[NT + 1];

    const float l0 = lab[row * 4 + 0], l1 = lab[row * 4 + 1];
    const float l2 = lab[row * 4 + 2], l3 = lab[row * 4 + 3];

    float lv[CHUNK];   // shifted logits for owned elements
    float qv[CHUNK];   // label-distance queries for owned elements

    // Load row, find max (diagonal included, matching reference).
    float m = -BIGF;
#pragma unroll
    for (int c = 0; c < CHUNK; ++c) {
        const int i = t + c * NT;
        const float v = logits[(size_t)row * NROWS + i];
        lv[c] = v;
        m = fmaxf(m, v);
    }
    s_red[t] = m;
    __syncthreads();
    for (int s = NT / 2; s > 0; s >>= 1) {
        if (t < s) s_red[t] = fmaxf(s_red[t], s_red[t + s]);
        __syncthreads();
    }
    const float rowmax = s_red[0];
    __syncthreads();

    // Shift, exponentiate, build label-distance keys. Diagonal -> key=BIG, exp=0.
#pragma unroll
    for (int c = 0; c < CHUNK; ++c) {
        const int i = t + c * NT;
        const float v = lv[c] - rowmax;
        lv[c] = v;
        if (i == row) {
            s_ex[i] = 0.0f; s_ld[i] = BIGF; qv[c] = BIGF;
        } else {
            s_ex[i] = expf(v);
            const float ld = fabsf(l0 - lab[i * 4 + 0]) + fabsf(l1 - lab[i * 4 + 1]) +
                             fabsf(l2 - lab[i * 4 + 2]) + fabsf(l3 - lab[i * 4 + 3]);
            s_ld[i] = ld; qv[c] = ld;
        }
    }
    __syncthreads();

    // Bitonic sort ascending by s_ld, carrying s_ex.
    for (int k = 2; k <= NROWS; k <<= 1) {
        for (int j = k >> 1; j > 0; j >>= 1) {
            for (int i = t; i < NROWS; i += NT) {
                const int ixj = i ^ j;
                if (ixj > i) {
                    const bool asc = ((i & k) == 0);
                    const float a = s_ld[i], b = s_ld[ixj];
                    if (asc ? (a > b) : (a < b)) {
                        s_ld[i] = b; s_ld[ixj] = a;
                        const float ea = s_ex[i];
                        s_ex[i] = s_ex[ixj]; s_ex[ixj] = ea;
                    }
                }
            }
            __syncthreads();
        }
    }

    // Inclusive scan of s_ex (chunked serial + cross-chunk exclusive scan).
    {
        const int base = t * CHUNK;
        float acc = 0.0f;
#pragma unroll
        for (int c = 0; c < CHUNK; ++c) { acc += s_ex[base + c]; s_ex[base + c] = acc; }
        s_red[t] = acc;
        __syncthreads();
        if (t == 0) {
            float run = 0.0f;
            for (int u = 0; u < NT; ++u) { const float v = s_red[u]; s_red[u] = run; run += v; }
            s_red[NT] = run;
        }
        __syncthreads();
        const float off = s_red[t];
#pragma unroll
        for (int c = 0; c < CHUNK; ++c) s_ex[base + c] += off;
    }
    const float total = s_red[NT];
    __syncthreads();

    // Lower-bound lookup per owned off-diagonal element; accumulate loss terms.
    float lsum = 0.0f;
#pragma unroll
    for (int c = 0; c < CHUNK; ++c) {
        const int i = t + c * NT;
        if (i == row) continue;
        const float q = qv[c];
        int lo = 0, hi = NROWS;
        while (lo < hi) {
            const int mid = (lo + hi) >> 1;
            if (s_ld[mid] < q) lo = mid + 1; else hi = mid;
        }
        const float below = (lo > 0) ? s_ex[lo - 1] : 0.0f;
        const float denom = total - below;
        lsum += lv[c] - logf(denom + LOSS_EPS);
    }
    __syncthreads();

    s_red[t] = lsum;
    __syncthreads();
    for (int s = NT / 2; s > 0; s >>= 1) {
        if (t < s) s_red[t] += s_red[t + s];
        __syncthreads();
    }
    if (t == 0) rowloss[row] = s_red[0];
}

// ---------------------------------------------------------------------------
// Pass 4: final deterministic reduction of per-row partials.
// ---------------------------------------------------------------------------
__global__ void rnc_final_kernel(const float* __restrict__ rowloss,
                                 float* __restrict__ out) {
    const int t = threadIdx.x;
    __shared__ float red[NT];
    float a = 0.0f;
    for (int i = t; i < NROWS; i += NT) a += rowloss[i];
    red[t] = a;
    __syncthreads();
    for (int s = NT / 2; s > 0; s >>= 1) {
        if (t < s) red[t] += red[t + s];
        __syncthreads();
    }
    if (t == 0) {
        const double denom = (double)NROWS * (double)(NROWS - 1);
        out[0] = (float)(-(double)red[0] / denom);
    }
}

// ---------------------------------------------------------------------------
extern "C" void kernel_launch(void* const* d_in, const int* in_sizes, int n_in,
                              void* d_out, int out_size, void* d_ws, size_t ws_size,
                              hipStream_t stream) {
    (void)in_sizes; (void)n_in; (void)out_size; (void)ws_size;

    const float* zi = (const float*)d_in[0];   // (2048, 256)
    const float* zj = (const float*)d_in[1];   // (2048, 256)
    const float* pi = (const float*)d_in[2];   // (2048, 4)
    const float* pj = (const float*)d_in[3];   // (2048, 4)

    char* ws = (char*)d_ws;
    // ws layout (all offsets 256B-aligned):
    //   [0,            2 MB)   fp16 normalized features (4096 x 256)
    //   [2 MB,        +64 KB)  labels f32 (4096 x 4)
    //   [.. ,         +16 KB)  rowloss f32 (4096)
    //   [.. ,         +64 MB)  logits f32 (4096 x 4096)
    _Float16* f16     = (_Float16*)(ws + 0);
    float*    lab     = (float*)(ws + 2097152);
    float*    rowloss = (float*)(ws + 2162688);
    float*    logits  = (float*)(ws + 2179072);

    rnc_prep_kernel<<<NROWS, NT, 0, stream>>>(zi, zj, pi, pj, f16, lab);

    const int n_tiles = (NROWS / 16) * (NROWS / 16);   // 65536 wave tiles
    rnc_gemm_kernel<<<n_tiles / 8, NT, 0, stream>>>(f16, logits);

    rnc_row_kernel<<<NROWS, NT, 0, stream>>>(logits, lab, rowloss);

    rnc_final_kernel<<<1, NT, 0, stream>>>(rowloss, (float*)d_out);
}
